// LFMAttentionBlock_81389630259296
// MI455X (gfx1250) — matrix operand order repaired and checked by reading, not compiled, not run on hardware
//
#include <hip/hip_runtime.h>
#include <hip/hip_bf16.h>

typedef __attribute__((ext_vector_type(16))) _Float16 v16h;
typedef __attribute__((ext_vector_type(8)))  _Float16 v8h;
typedef __attribute__((ext_vector_type(8)))  float    v8f;
typedef __attribute__((ext_vector_type(4)))  unsigned v4u;
typedef __attribute__((ext_vector_type(8)))  unsigned v8u;

#define B_   2
#define S_   2048
#define H_   768
#define NH_  12
#define NKV_ 4
#define HD_  64
#define M_   (B_*S_)

// ---------------------------------------------------------------------------
// CDNA5 async global->LDS copy (ASYNCcnt-tracked)
// ---------------------------------------------------------------------------
__device__ __forceinline__ void async_b128(void* lds, const void* g) {
    unsigned l = (unsigned)(unsigned long long)lds;          // flat->LDS low32
    unsigned long long a = (unsigned long long)g;
    asm volatile("global_load_async_to_lds_b128 %0, %1, off"
                 :: "v"(l), "v"(a) : "memory");
}
template <int N>
__device__ __forceinline__ void wait_asynccnt() {
    asm volatile("s_wait_asynccnt %0" :: "n"(N) : "memory");
}

// ---------------------------------------------------------------------------
// CDNA5 Tensor Data Mover: 2D tile load global->LDS (TENSORcnt-tracked).
// D# per CDNA5 ISA: group0 (4 SGPRs) + group1 (8 SGPRs).
// ---------------------------------------------------------------------------
__device__ __forceinline__ void tdm_load_2d(unsigned lds_addr, const void* gaddr,
                                            unsigned tdim0, unsigned tdim1,
                                            unsigned tile0, unsigned tile1,
                                            unsigned stride0_elems) {
    unsigned long long ga = (unsigned long long)gaddr;
    v4u g0;
    g0[0] = 1u;                                               // count=1 (valid)
    g0[1] = lds_addr;                                         // lds_addr
    g0[2] = (unsigned)ga;                                     // global_addr lo
    g0[3] = ((unsigned)(ga >> 32) & 0x01FFFFFFu) | 0x80000000u; // hi | type=2
    v8u g1;
    g1[0] = 0x10000u;                                         // data_size=1 (2B)
    g1[1] = (tdim0 & 0xFFFFu) << 16;                          // tensor_dim0 lo
    g1[2] = (tdim0 >> 16) | ((tdim1 & 0xFFFFu) << 16);        // dim0 hi|dim1 lo
    g1[3] = (tdim1 >> 16) | (tile0 << 16);                    // dim1 hi|tile0
    g1[4] = tile1;                                            // tile1 | tile2=0
    g1[5] = stride0_elems;                                    // dim0_stride lo32
    g1[6] = 0u;                                               // stride hi, etc.
    g1[7] = 0u;
    asm volatile("tensor_load_to_lds %0, %1" :: "s"(g0), "s"(g1) : "memory");
}

// ---------------------------------------------------------------------------
// WMMA helpers (CDNA5 16x16x32 f16 -> f32)
// ---------------------------------------------------------------------------
__device__ __forceinline__ v8f wmma_f16(v16h a, v16h b, v8f c) {
    return __builtin_amdgcn_wmma_f32_16x16x32_f16(
        false, a, false, b, (short)0, c, false, false);
}

// A fragment: 16x32 f16, element (m,k) at p[m*stride + k], m = lane&15.
__device__ __forceinline__ v16h frag_a(const _Float16* p, int stride, int kbase) {
    int lane = threadIdx.x & 31;
    int m  = lane & 15;
    int hi = lane >> 4;
    const _Float16* rp = p + (size_t)m * stride + kbase;
    v8h lo = *(const v8h*)(rp + 8 * hi);
    v8h hg = *(const v8h*)(rp + 16 + 8 * hi);
    v16h a;
#pragma unroll
    for (int i = 0; i < 8; ++i) { a[i] = lo[i]; a[i + 8] = hg[i]; }
    return a;
}

// B fragment: 32x16 f16, element (k,n) at p[n*stride + k] (K-contiguous).
__device__ __forceinline__ v16h frag_b_t(const _Float16* p, int stride, int kbase) {
    int lane = threadIdx.x & 31;
    int n  = lane & 15;
    int hi = lane >> 4;
    const _Float16* cp = p + (size_t)n * stride + kbase + 8 * hi;
    v8h lo = *(const v8h*)(cp);
    v8h hg = *(const v8h*)(cp + 16);
    v16h b;
#pragma unroll
    for (int i = 0; i < 8; ++i) { b[i] = lo[i]; b[i + 8] = hg[i]; }
    return b;
}

// ---------------------------------------------------------------------------
// Weight convert + transpose: src f32 [K,N] -> dst f16 [N,K]
// ---------------------------------------------------------------------------
__global__ __launch_bounds__(256) void cvt_f16_t(const float* __restrict__ src,
                                                 _Float16* __restrict__ dst,
                                                 int K, int N) {
    int idx = blockIdx.x * 256 + threadIdx.x;           // write-coalesced in k
    int k = idx % K;
    int n = idx / K;
    if (n < N) dst[(size_t)n * K + k] = (_Float16)src[(size_t)k * N + n];
}

// ---------------------------------------------------------------------------
// Fused LayerNorm -> f16 (one block per row of 768)
// ---------------------------------------------------------------------------
__global__ __launch_bounds__(256) void ln_f16(const float* __restrict__ X,
                                              const float* __restrict__ gamma,
                                              const float* __restrict__ beta,
                                              _Float16* __restrict__ O) {
    __shared__ float red[16];
    int row = blockIdx.x;
    int tid = threadIdx.x;
    const float* xr = X + (size_t)row * H_;
    float x[3], s = 0.f, ss = 0.f;
#pragma unroll
    for (int i = 0; i < 3; ++i) {
        x[i] = xr[tid + i * 256];
        s += x[i]; ss += x[i] * x[i];
    }
#pragma unroll
    for (int off = 1; off < 32; off <<= 1) {
        s  += __shfl_xor(s,  off);
        ss += __shfl_xor(ss, off);
    }
    int wid = tid >> 5, lane = tid & 31;
    if (lane == 0) { red[wid] = s; red[8 + wid] = ss; }
    __syncthreads();
    if (tid == 0) {
        float ts = 0.f, tss = 0.f;
#pragma unroll
        for (int w = 0; w < 8; ++w) { ts += red[w]; tss += red[8 + w]; }
        red[0] = ts; red[1] = tss;
    }
    __syncthreads();
    float mu   = red[0] * (1.f / H_);
    float var  = red[1] * (1.f / H_) - mu * mu;
    float rstd = rsqrtf(var + 1e-12f);
#pragma unroll
    for (int i = 0; i < 3; ++i) {
        int c = tid + i * 256;
        float v = (x[i] - mu) * rstd * gamma[c] + beta[c];
        O[(size_t)row * H_ + c] = (_Float16)v;
    }
}

// ---------------------------------------------------------------------------
// Tiled WMMA GEMM: C[M,N] = A[M,K](f16) * Wt[N,K](f16)^T + bias (+ residual)
// Block tile 128x128, 8 waves (4Mx2N), 32Mx64N per wave.
// LDS tiles double-buffered, staged by the Tensor Data Mover (one
// tensor_load_to_lds per tile, issued by wave 0, TENSORcnt-pipelined).
// ---------------------------------------------------------------------------
__global__ __launch_bounds__(256) void gemm_f16(const _Float16* __restrict__ A,
                                                const _Float16* __restrict__ Wt,
                                                const float* __restrict__ bias,
                                                const float* __restrict__ residual,
                                                float* __restrict__ C,
                                                int Mdim, int Ndim, int Kdim) {
    __shared__ __align__(16) _Float16 lA[2][128 * 32];
    __shared__ __align__(16) _Float16 lB[2][128 * 32];
    int tid  = threadIdx.x;
    int wid  = tid >> 5, lane = tid & 31;
    int hi   = lane >> 4, nl = lane & 15;
    int m0   = blockIdx.x * 128;
    int n0   = blockIdx.y * 128;
    int wy   = wid >> 1, wx = wid & 1;

    const _Float16* Ab = A  + (size_t)m0 * Kdim;   // A tile rows base
    const _Float16* Wb = Wt + (size_t)n0 * Kdim;   // B tile rows base

    auto stage = [&](int buf, int k0) {            // wave 0 drives the TDM
        if (wid == 0) {
            tdm_load_2d((unsigned)(unsigned long long)(lA[buf]), Ab + k0,
                        (unsigned)Kdim, (unsigned)Mdim, 32u, 128u, (unsigned)Kdim);
            tdm_load_2d((unsigned)(unsigned long long)(lB[buf]), Wb + k0,
                        (unsigned)Kdim, (unsigned)Ndim, 32u, 128u, (unsigned)Kdim);
        }
    };

    v8f acc[2][4] = {};
    auto compute = [&](int cur) {
        v16h a0 = frag_a(lA[cur] + (32 * wy) * 32, 32, 0);
        v16h a1 = frag_a(lA[cur] + (32 * wy + 16) * 32, 32, 0);
#pragma unroll
        for (int nt = 0; nt < 4; ++nt) {
            v16h b = frag_b_t(lB[cur] + (64 * wx + nt * 16) * 32, 32, 0);
            acc[0][nt] = wmma_f16(a0, b, acc[0][nt]);
            acc[1][nt] = wmma_f16(a1, b, acc[1][nt]);
        }
    };

    stage(0, 0);
    int k0 = 0;
    for (; k0 < Kdim - 32; k0 += 32) {             // steady state: no branches
        int cur = (k0 >> 5) & 1;
        stage(cur ^ 1, k0 + 32);
        if (wid == 0) __builtin_amdgcn_s_wait_tensorcnt((short)2);
        __syncthreads();
        compute(cur);
        __syncthreads();
    }
    if (wid == 0) __builtin_amdgcn_s_wait_tensorcnt((short)0);
    __syncthreads();
    compute((k0 >> 5) & 1);

#pragma unroll
    for (int nt = 0; nt < 4; ++nt) {
        int n = n0 + 64 * wx + nt * 16 + nl;
        float bval = bias[n];
#pragma unroll
        for (int mt = 0; mt < 2; ++mt) {
#pragma unroll
            for (int r = 0; r < 8; ++r) {
                int m = m0 + 32 * wy + 16 * mt + r + 8 * hi;
                size_t off = (size_t)m * Ndim + n;
                float v = acc[mt][nt][r] + bval;
                if (residual) v += residual[off];
                C[off] = v;
            }
        }
    }
}

// ---------------------------------------------------------------------------
// RoPE + relayout: x32 [B,S,NHx,HD] f32  ->  x16 [B,NHx,S,HD] f16
// ---------------------------------------------------------------------------
__global__ __launch_bounds__(256) void rope_relayout(const float* __restrict__ x32,
                                                     _Float16* __restrict__ x16,
                                                     int NHx) {
    int idx = blockIdx.x * 256 + threadIdx.x;
    int j = idx & 31;
    int t = idx >> 5;
    int h = t % NHx;  t /= NHx;
    int s = t % S_;
    int b = t / S_;
    float freq = __expf(-(float)j * (2.f / HD_) * 9.210340371976184f);
    float cs, sn;
    __sincosf((float)s * freq, &sn, &cs);
    size_t in  = (((size_t)(b * S_ + s)) * NHx + h) * HD_;
    size_t out = (((size_t)(b * NHx + h)) * S_ + s) * HD_;
    float x1 = x32[in + j];
    float x2 = x32[in + j + 32];
    x16[out + j]      = (_Float16)(x1 * cs - x2 * sn);
    x16[out + j + 32] = (_Float16)(x1 * sn + x2 * cs);
}

// V relayout+transpose: v32 [B,S,NKV,HD] f32 -> vt16 [B,NKV,HD,S] f16
__global__ __launch_bounds__(256) void vt_relayout(const float* __restrict__ v32,
                                                   _Float16* __restrict__ vt16) {
    int idx = blockIdx.x * 256 + threadIdx.x;       // write-coalesced in s
    int s = idx & (S_ - 1);
    int t = idx >> 11;
    int d = t % HD_;  t /= HD_;
    int h = t % NKV_;
    int b = t / NKV_;
    vt16[(((size_t)(b * NKV_ + h)) * HD_ + d) * S_ + s] =
        (_Float16)v32[(((size_t)(b * S_ + s)) * NKV_ + h) * HD_ + d];
}

// ---------------------------------------------------------------------------
// Flash attention (GQA). grid = (B*NH, S/128), 8 waves/block, 16 queries/wave.
// K [key][d] and Vt [d][key] chunks (32 keys) double-buffered via ASYNCcnt
// global->LDS copies.
// ---------------------------------------------------------------------------
__global__ __launch_bounds__(256) void attn_fwd(const _Float16* __restrict__ Q,
                                                const _Float16* __restrict__ K,
                                                const _Float16* __restrict__ Vt,
                                                _Float16* __restrict__ O) {
    __shared__ __align__(16) _Float16 lK[2][32 * 64];
    __shared__ __align__(16) _Float16 lV[2][64 * 32];
    __shared__ __align__(16) _Float16 lP[8 * 16 * 32];
    int tid  = threadIdx.x;
    int wid  = tid >> 5, lane = tid & 31;
    int hi   = lane >> 4, nl = lane & 15;
    int bh   = blockIdx.x;
    int b    = bh / NH_;
    int h    = bh % NH_;
    int kvh  = h / (NH_ / NKV_);
    const _Float16* Qh  = Q  + ((size_t)(b * NH_  + h))   * S_ * HD_;
    const _Float16* Kh  = K  + ((size_t)(b * NKV_ + kvh)) * S_ * HD_;
    const _Float16* Vth = Vt + ((size_t)(b * NKV_ + kvh)) * HD_ * S_;
    int q0 = (blockIdx.y * 8 + wid) * 16;

    v16h aq0 = frag_a(Qh + (size_t)q0 * HD_, HD_, 0);
    v16h aq1 = frag_a(Qh + (size_t)q0 * HD_, HD_, 32);

    int krow = tid >> 3, kch = (tid & 7) * 8;     // K tile: 32 x 64
    int vrow = tid >> 2, vch = (tid & 3) * 8;     // Vt tile: 64 x 32
    auto stage = [&](int buf, int kb) {
        async_b128(lK[buf] + krow * 64 + kch, Kh  + (size_t)(kb + krow) * HD_ + kch);
        async_b128(lV[buf] + vrow * 32 + vch, Vth + (size_t)vrow * S_ + kb + vch);
    };

    float mrow[8], lrow[8];
    v8f o[4] = {};
#pragma unroll
    for (int r = 0; r < 8; ++r) { mrow[r] = -1e30f; lrow[r] = 0.f; }

    auto process = [&](int cur) {
        v8f s0 = {}, s1 = {};
        s0 = wmma_f16(aq0, frag_b_t(lK[cur], 64, 0),  s0);
        s0 = wmma_f16(aq1, frag_b_t(lK[cur], 64, 32), s0);
        s1 = wmma_f16(aq0, frag_b_t(lK[cur] + 16 * 64, 64, 0),  s1);
        s1 = wmma_f16(aq1, frag_b_t(lK[cur] + 16 * 64, 64, 32), s1);

        const float sc = 0.125f;  // 1/sqrt(64)
        _Float16* lp = lP + wid * 16 * 32;
#pragma unroll
        for (int r = 0; r < 8; ++r) {
            float a0 = s0[r] * sc, a1 = s1[r] * sc;
            float vmax = fmaxf(a0, a1);
#pragma unroll
            for (int off = 1; off < 16; off <<= 1)
                vmax = fmaxf(vmax, __shfl_xor(vmax, off, 16));
            float mnew  = fmaxf(mrow[r], vmax);
            float alpha = __expf(mrow[r] - mnew);
            float p0 = __expf(a0 - mnew);
            float p1 = __expf(a1 - mnew);
            float ps = p0 + p1;
#pragma unroll
            for (int off = 1; off < 16; off <<= 1)
                ps += __shfl_xor(ps, off, 16);
            lrow[r] = lrow[r] * alpha + ps;
            mrow[r] = mnew;
#pragma unroll
            for (int nt = 0; nt < 4; ++nt) o[nt][r] *= alpha;
            int m = r + 8 * hi;
            lp[m * 32 + nl]      = (_Float16)p0;
            lp[m * 32 + 16 + nl] = (_Float16)p1;
        }
        asm volatile("s_wait_dscnt 0x0" ::: "memory");

        v16h ap = frag_a(lp, 32, 0);
#pragma unroll
        for (int nt = 0; nt < 4; ++nt)
            o[nt] = wmma_f16(ap, frag_b_t(lV[cur] + nt * 16 * 32, 32, 0), o[nt]);
    };

    stage(0, 0);
    int kb = 0;
    for (; kb < S_ - 32; kb += 32) {               // steady state: no branches
        int cur = (kb >> 5) & 1;
        stage(cur ^ 1, kb + 32);
        wait_asynccnt<2>();
        __syncthreads();
        process(cur);
        __syncthreads();
    }
    wait_asynccnt<0>();
    __syncthreads();
    process((kb >> 5) & 1);

#pragma unroll
    for (int nt = 0; nt < 4; ++nt) {
#pragma unroll
        for (int r = 0; r < 8; ++r) {
            int m = q0 + r + 8 * hi;
            int d = nt * 16 + nl;
            float val = o[nt][r] / lrow[r];
            O[((size_t)(b * S_ + m)) * H_ + h * HD_ + d] = (_Float16)val;
        }
    }
}

// ---------------------------------------------------------------------------
// Host-side launch
// ---------------------------------------------------------------------------
extern "C" void kernel_launch(void* const* d_in, const int* in_sizes, int n_in,
                              void* d_out, int out_size, void* d_ws, size_t ws_size,
                              hipStream_t stream) {
    (void)in_sizes; (void)n_in; (void)out_size; (void)ws_size;
    const float* hs    = (const float*)d_in[0];
    const float* Wq    = (const float*)d_in[1];
    const float* bq    = (const float*)d_in[2];
    const float* Wk    = (const float*)d_in[3];
    const float* bk    = (const float*)d_in[4];
    const float* Wv    = (const float*)d_in[5];
    const float* bv    = (const float*)d_in[6];
    const float* Wo    = (const float*)d_in[7];
    const float* bo    = (const float*)d_in[8];
    const float* gamma = (const float*)d_in[9];
    const float* beta  = (const float*)d_in[10];
    float* out = (float*)d_out;

    char* w = (char*)d_ws;
    size_t off = 0;
    auto alloc = [&](size_t bytes) {
        void* p = w + off;
        off = (off + bytes + 255) & ~(size_t)255;
        return p;
    };
    _Float16* h16   = (_Float16*)alloc((size_t)M_ * H_ * 2);
    _Float16* wq16  = (_Float16*)alloc((size_t)H_ * H_ * 2);            // [N,K]
    _Float16* wk16  = (_Float16*)alloc((size_t)H_ * NKV_ * HD_ * 2);
    _Float16* wv16  = (_Float16*)alloc((size_t)H_ * NKV_ * HD_ * 2);
    _Float16* wo16  = (_Float16*)alloc((size_t)H_ * H_ * 2);
    float*    q32   = (float*)alloc((size_t)M_ * H_ * 4);
    float*    k32   = (float*)alloc((size_t)M_ * NKV_ * HD_ * 4);
    float*    v32   = (float*)alloc((size_t)M_ * NKV_ * HD_ * 4);
    _Float16* q16   = (_Float16*)alloc((size_t)B_ * NH_ * S_ * HD_ * 2);
    _Float16* k16   = (_Float16*)alloc((size_t)B_ * NKV_ * S_ * HD_ * 2);
    _Float16* vt16  = (_Float16*)alloc((size_t)B_ * NKV_ * HD_ * S_ * 2);
    _Float16* at16  = (_Float16*)alloc((size_t)M_ * H_ * 2);

    // 1) weights -> f16, transposed to [N,K]
    cvt_f16_t<<<(H_ * H_) / 256, 256, 0, stream>>>(Wq, wq16, H_, H_);
    cvt_f16_t<<<(H_ * NKV_ * HD_) / 256, 256, 0, stream>>>(Wk, wk16, H_, NKV_ * HD_);
    cvt_f16_t<<<(H_ * NKV_ * HD_) / 256, 256, 0, stream>>>(Wv, wv16, H_, NKV_ * HD_);
    cvt_f16_t<<<(H_ * H_) / 256, 256, 0, stream>>>(Wo, wo16, H_, H_);

    // 2) LayerNorm -> f16
    ln_f16<<<M_, 256, 0, stream>>>(hs, gamma, beta, h16);

    // 3) Q/K/V projections (WMMA GEMM, TDM double-buffered)
    gemm_f16<<<dim3(M_ / 128, H_ / 128), 256, 0, stream>>>(
        h16, wq16, bq, nullptr, q32, M_, H_, H_);
    gemm_f16<<<dim3(M_ / 128, (NKV_ * HD_) / 128), 256, 0, stream>>>(
        h16, wk16, bk, nullptr, k32, M_, NKV_ * HD_, H_);
    gemm_f16<<<dim3(M_ / 128, (NKV_ * HD_) / 128), 256, 0, stream>>>(
        h16, wv16, bv, nullptr, v32, M_, NKV_ * HD_, H_);

    // 4) RoPE + head-major relayout; V transposed per head
    rope_relayout<<<(B_ * S_ * NH_ * 32) / 256, 256, 0, stream>>>(q32, q16, NH_);
    rope_relayout<<<(B_ * S_ * NKV_ * 32) / 256, 256, 0, stream>>>(k32, k16, NKV_);
    vt_relayout<<<(B_ * S_ * NKV_ * HD_) / 256, 256, 0, stream>>>(v32, vt16);

    // 5) flash attention
    attn_fwd<<<dim3(B_ * NH_, S_ / 128), 256, 0, stream>>>(q16, k16, vt16, at16);

    // 6) output projection + residual -> d_out
    gemm_f16<<<dim3(M_ / 128, H_ / 128), 256, 0, stream>>>(
        at16, wo16, bo, hs, out, M_, H_, H_);
}
